// GCN_JK_11424613007585
// MI455X (gfx1250) — matrix operand order repaired and checked by reading, not compiled, not run on hardware
//
#include <hip/hip_runtime.h>

typedef __attribute__((ext_vector_type(16))) _Float16 v16h;
typedef __attribute__((ext_vector_type(8)))  _Float16 v8h;
typedef __attribute__((ext_vector_type(8)))  float    v8f;

#define HIDC 256

// ---------------- small utility kernels ----------------

__global__ void k_deg_init(float* __restrict__ deg, int n) {
  int i = blockIdx.x * blockDim.x + threadIdx.x;
  if (i < n) deg[i] = 1.0f;                 // self loop contributes 1 to degree
}

__global__ void k_deg_edges(const int* __restrict__ dst, float* __restrict__ deg, int e) {
  int i = blockIdx.x * blockDim.x + threadIdx.x;
  if (i < e) unsafeAtomicAdd(&deg[dst[i]], 1.0f);   // -> global_atomic_add_f32
}

__global__ void k_dinv(float* __restrict__ d, int n) {
  int i = blockIdx.x * blockDim.x + threadIdx.x;
  if (i < n) d[i] = rsqrtf(d[i]);           // deg >= 1 always (self loops)
}

// Pre-swizzle W[256,Kout] (f32) into WMMA-B fragment order (f16):
//   Wsw[ (g*Kout + n)*16 + e ] = W[ k*Kout + n ], k = kb*32 + 8*half + (e&7) + 16*(e>>3), g = kb*2+half
// so each lane's 16-element B fragment is one contiguous, 32B-aligned chunk.
__global__ void k_w_swizzle(const float* __restrict__ W, _Float16* __restrict__ out, int Kout) {
  int o = blockIdx.x * blockDim.x + threadIdx.x;
  if (o >= HIDC * Kout) return;
  const int e = o & 15;
  const int rest = o >> 4;
  const int n = rest % Kout;
  const int g = rest / Kout;                // 0..15  (= kb*2 + half)
  const int k = (g >> 1) * 32 + (g & 1) * 8 + (e & 7) + (e >> 3) * 16;  // kb*32 + 8*half + e (e<8), +16 for e>=8
  out[o] = (_Float16)W[(size_t)k * Kout + n];
}

// ---------------- WMMA GEMM: C[N,KOUT] = A[N,256] @ W[256,KOUT] (+bias) ----------------
// Block = 128 threads = 4 waves; block computes rows [16*bx, +16) and the FULL output
// width: wave w owns NT 16x16 tiles. All loop offsets are compile-time immediates:
// one base pointer per lane, then immediate-offset global_load_b128 / ds_load_b128.

template <int NT, int KOUT, bool BIAS, bool AUX>
__global__ __launch_bounds__(128)
void k_gemm_wmma(const float* __restrict__ A, const _Float16* __restrict__ Wsw,
                 const float* __restrict__ bias, float* __restrict__ C,
                 float* __restrict__ aux, int auxLd, int auxOff) {
  __shared__ _Float16 As[16][HIDC + 16];    // row stride 544B (16B multiple)

  const int r0 = blockIdx.x * 16;
  const int t  = threadIdx.x;

  // Cooperative coalesced load of the 16x256 A tile: float4 global loads,
  // f32->f16 convert, contiguous 8B stores to LDS.
#pragma unroll
  for (int i = 0; i < 8; ++i) {
    const int idx4 = i * 128 + t;           // 0..1023 float4 chunks
    const int row = idx4 >> 6;
    const int col = (idx4 & 63) * 4;
    const float4 v = *(const float4*)(A + (unsigned)(r0 + row) * HIDC + col);
    As[row][col + 0] = (_Float16)v.x;
    As[row][col + 1] = (_Float16)v.y;
    As[row][col + 2] = (_Float16)v.z;
    As[row][col + 3] = (_Float16)v.w;
  }
  __syncthreads();

  const int wave  = t >> 5;
  const int lane  = t & 31;
  const int half  = lane >> 4;
  const int m0    = lane & 15;
  const int nbase = wave * (16 * NT) + (lane & 15);

  // One base pointer per lane; everything else is a compile-time immediate.
  const _Float16* __restrict__ wp = Wsw + ((unsigned)(half * KOUT + nbase) << 4);
  const _Float16* __restrict__ ap = &As[m0][8 * half];

  v8f acc[NT];
#pragma unroll
  for (int ti = 0; ti < NT; ++ti) acc[ti] = (v8f){};

#pragma unroll
  for (int kb = 0; kb < HIDC / 32; ++kb) {
    // A fragment: elements 0-7 = K [kb*32+8*half, +8), 8-15 = +16 (exact ISA order)
    v16h afrag;
    ((v8h*)&afrag)[0] = *(const v8h*)(ap + kb * 32);
    ((v8h*)&afrag)[1] = *(const v8h*)(ap + kb * 32 + 16);

#pragma unroll
    for (int ti = 0; ti < NT; ++ti) {
      const v16h bfrag = *(const v16h*)(wp + kb * (2 * KOUT * 16) + ti * 256);
      acc[ti] = __builtin_amdgcn_wmma_f32_16x16x32_f16(
          false, afrag, false, bfrag, (short)0, acc[ti], false, false);
    }
  }

  // Epilogue: 32-bit index math, immediate-offset stores.
  const unsigned cbase = (unsigned)(r0 + 8 * half) * KOUT + nbase;
  const unsigned abase = AUX ? ((unsigned)(r0 + 8 * half) * auxLd + auxOff + nbase) : 0u;
#pragma unroll
  for (int ti = 0; ti < NT; ++ti) {
    const float bv = BIAS ? bias[nbase + ti * 16] : 0.0f;
#pragma unroll
    for (int r = 0; r < 8; ++r) {
      const float v = acc[ti][r] + bv;
      C[cbase + r * KOUT + ti * 16] = v;
      if (AUX) aux[abase + (unsigned)r * auxLd + ti * 16] = v;
    }
  }
}

// ---------------- GCN aggregation (CF = channel count, compile-time) ----------------

// agg[i,c] = hlin[i,c] * dinv[i]^2 + bias[c]   (self-loop term + bias)
template <int CF>
__global__ void k_agg_init(const float* __restrict__ hlin, const float* __restrict__ dinv,
                           const float* __restrict__ bias, float* __restrict__ agg, int N) {
  constexpr int CQ = CF >> 2;
  int tid = blockIdx.x * blockDim.x + threadIdx.x;
  if (tid >= N * CQ) return;
  const unsigned i = (unsigned)tid / CQ;
  const unsigned c = ((unsigned)tid % CQ) * 4;
  float w = dinv[i]; w *= w;
  const float4 h = *(const float4*)(hlin + i * CF + c);
  const float4 b = *(const float4*)(bias + c);
  float4 r;
  r.x = h.x * w + b.x; r.y = h.y * w + b.y; r.z = h.z * w + b.z; r.w = h.w * w + b.w;
  *(float4*)(agg + i * CF + c) = r;
}

// agg[dst,c] += hlin[src,c] * dinv[src]*dinv[dst]  (edge-parallel, float4 channels)
template <int CF>
__global__ void k_edge_agg(const int* __restrict__ src, const int* __restrict__ dst,
                           const float* __restrict__ dinv, const float* __restrict__ hlin,
                           float* __restrict__ agg, int E) {
  constexpr int CQ = CF >> 2;
  int tid = blockIdx.x * blockDim.x + threadIdx.x;
  if (tid >= E * CQ) return;
  const unsigned e = (unsigned)tid / CQ;
  const unsigned c = ((unsigned)tid % CQ) * 4;
  const int s = src[e], d = dst[e];
  const float nm = dinv[s] * dinv[d];
  const float4 h = *(const float4*)(hlin + (unsigned)s * CF + c);
  float* ap = agg + (unsigned)d * CF + c;
  unsafeAtomicAdd(ap + 0, h.x * nm);
  unsafeAtomicAdd(ap + 1, h.y * nm);
  unsafeAtomicAdd(ap + 2, h.z * nm);
  unsafeAtomicAdd(ap + 3, h.w * nm);
}

// hnext = relu?(agg);  out[:, off:off+CF] = hnext   (JumpingKnowledge concat)
template <int CF, bool RELU, bool NEXT>
__global__ void k_finalize(const float* __restrict__ agg, float* __restrict__ hnext,
                           float* __restrict__ out, int ldo, int off, int N) {
  constexpr int CQ = CF >> 2;
  int tid = blockIdx.x * blockDim.x + threadIdx.x;
  if (tid >= N * CQ) return;
  const unsigned i = (unsigned)tid / CQ;
  const unsigned c = ((unsigned)tid % CQ) * 4;
  float4 v = *(const float4*)(agg + i * CF + c);
  if (RELU) {
    v.x = fmaxf(v.x, 0.f); v.y = fmaxf(v.y, 0.f);
    v.z = fmaxf(v.z, 0.f); v.w = fmaxf(v.w, 0.f);
  }
  if (NEXT) *(float4*)(hnext + i * CF + c) = v;
  *(float4*)(out + i * (unsigned)ldo + off + c) = v;
}

// ---------------- host orchestration ----------------

extern "C" void kernel_launch(void* const* d_in, const int* in_sizes, int n_in,
                              void* d_out, int out_size, void* d_ws, size_t ws_size,
                              hipStream_t stream) {
  (void)n_in; (void)out_size; (void)ws_size;
  const float* x   = (const float*)d_in[0];
  const int*   ei  = (const int*)d_in[1];
  const float* We  = (const float*)d_in[2];
  const float* be  = (const float*)d_in[3];
  const float* W1  = (const float*)d_in[4];
  const float* b1  = (const float*)d_in[5];
  const float* W2  = (const float*)d_in[6];
  const float* b2  = (const float*)d_in[7];
  const float* W3  = (const float*)d_in[8];
  const float* b3  = (const float*)d_in[9];
  const float* W4  = (const float*)d_in[10];
  const float* b4  = (const float*)d_in[11];

  const int N    = in_sizes[0] / HIDC;      // 50000 (multiple of 16)
  const int E    = in_sizes[1] / 2;         // 800000
  const int NCLS = in_sizes[10] / HIDC;     // 128
  const int ldo  = 4 * HIDC + NCLS;         // 1152
  const int* srcI = ei;
  const int* dstI = ei + E;
  float* out = (float*)d_out;

  // workspace carve-out (~155 MB)
  char* ws = (char*)d_ws;
  size_t cur = 0;
  auto alloc = [&](size_t bytes) -> void* {
    void* p = ws + cur;
    cur += (bytes + 255) & ~(size_t)255;
    return p;
  };
  float*    dinv = (float*)alloc((size_t)N * 4);
  float*    hcur = (float*)alloc((size_t)N * HIDC * 4);
  float*    hlin = (float*)alloc((size_t)N * HIDC * 4);
  float*    agg  = (float*)alloc((size_t)N * HIDC * 4);
  _Float16* hWe  = (_Float16*)alloc((size_t)in_sizes[2] * 2);
  _Float16* hW1  = (_Float16*)alloc((size_t)in_sizes[4] * 2);
  _Float16* hW2  = (_Float16*)alloc((size_t)in_sizes[6] * 2);
  _Float16* hW3  = (_Float16*)alloc((size_t)in_sizes[8] * 2);
  _Float16* hW4  = (_Float16*)alloc((size_t)in_sizes[10] * 2);

  const int TB = 256;
  auto blocks = [&](long long w) { return (int)((w + TB - 1) / TB); };

  // 1) weights -> f16 in WMMA-B fragment order
  k_w_swizzle<<<blocks(in_sizes[2]),  TB, 0, stream>>>(We, hWe, HIDC);
  k_w_swizzle<<<blocks(in_sizes[4]),  TB, 0, stream>>>(W1, hW1, HIDC);
  k_w_swizzle<<<blocks(in_sizes[6]),  TB, 0, stream>>>(W2, hW2, HIDC);
  k_w_swizzle<<<blocks(in_sizes[8]),  TB, 0, stream>>>(W3, hW3, HIDC);
  k_w_swizzle<<<blocks(in_sizes[10]), TB, 0, stream>>>(W4, hW4, NCLS);

  // 2) GCN normalization: deg -> dinv
  k_deg_init <<<blocks(N), TB, 0, stream>>>(dinv, N);
  k_deg_edges<<<blocks(E), TB, 0, stream>>>(dstI, dinv, E);
  k_dinv     <<<blocks(N), TB, 0, stream>>>(dinv, N);

  // 3) h0 = x @ W_embed + b_embed  (also written to out[:,0:256])
  dim3 gH(N / 16, 1);
  k_gemm_wmma<4, HIDC, true, true><<<gH, 128, 0, stream>>>(x, hWe, be, hcur, out, ldo, 0);

  // 4) three GCN conv layers with ReLU
  const _Float16* Wl[3] = {hW1, hW2, hW3};
  const float*    bl[3] = {b1, b2, b3};
  const long long workN = (long long)N * (HIDC / 4);
  const long long workE = (long long)E * (HIDC / 4);
  for (int l = 0; l < 3; ++l) {
    k_gemm_wmma<4, HIDC, false, false><<<gH, 128, 0, stream>>>(hcur, Wl[l], nullptr, hlin,
                                                               nullptr, 0, 0);
    k_agg_init<HIDC><<<blocks(workN), TB, 0, stream>>>(hlin, dinv, bl[l], agg, N);
    k_edge_agg<HIDC><<<blocks(workE), TB, 0, stream>>>(srcI, dstI, dinv, hlin, agg, E);
    k_finalize<HIDC, true, true><<<blocks(workN), TB, 0, stream>>>(agg, hcur, out, ldo,
                                                                   (l + 1) * HIDC, N);
  }

  // 5) final GCN conv (no ReLU) into out[:,1024:1152]
  const long long workN4 = (long long)N * (128 / 4);
  const long long workE4 = (long long)E * (128 / 4);
  k_gemm_wmma<2, 128, false, false><<<gH, 128, 0, stream>>>(hcur, hW4, nullptr, hlin,
                                                            nullptr, 0, 0);
  k_agg_init<128><<<blocks(workN4), TB, 0, stream>>>(hlin, dinv, b4, agg, N);
  k_edge_agg<128><<<blocks(workE4), TB, 0, stream>>>(srcI, dstI, dinv, hlin, agg, E);
  k_finalize<128, false, false><<<blocks(workN4), TB, 0, stream>>>(agg, nullptr, out, ldo,
                                                                   4 * HIDC, N);
}
